// ViSNetBlock_57904749085233
// MI455X (gfx1250) — matrix operand order repaired and checked by reading, not compiled, not run on hardware
//
#include <hip/hip_runtime.h>
#include <cstdint>
#include <cstddef>

// ---------------- problem constants (from reference) ----------------
#define NN 10000
#define EE 64000
#define HH 128
#define LL 8

typedef __bf16 bf16_t;
typedef __attribute__((ext_vector_type(16))) __bf16 v16bf;
typedef __attribute__((ext_vector_type(8)))  float  v8f;

// ---------------- WMMA helpers (CDNA5 gfx1250, wave32) ----------------
__device__ __forceinline__ v8f wmma_bf16(v16bf a, v16bf b, v8f c) {
  // D(f32 16x16) = A(bf16 16x32) * B(bf16 32x16) + C
  return __builtin_amdgcn_wmma_f32_16x16x32_bf16(
      /*neg_a=*/false, a, /*neg_b=*/false, b,
      /*c_mod=*/(short)0, c, /*reuse_a=*/false, /*reuse_b=*/false);
}

// A-matrix per-lane fragment: lane<16 -> K = k0+[0..15]
//                             lane>=16 -> K = k0+[16..31]
__device__ __forceinline__ v16bf load_a_row(const bf16_t* __restrict__ rowbase,
                                            int k0, int lane) {
  int kb = k0 + ((lane & 16) ? 16 : 0);
  union { v16bf v; uint4 u[2]; } r;
  r.u[0] = *reinterpret_cast<const uint4*>(rowbase + kb);
  r.u[1] = *reinterpret_cast<const uint4*>(rowbase + kb + 8);
  return r.v;
}

// B fragments are pre-packed: pack[kt][ntile][lane][16 bf16] where lane's 16
// values are B[kt*32 + (lane<16?0:16) + i][ntile*16 + lane%16], i=0..15.
__device__ __forceinline__ v16bf load_b_pack(const bf16_t* __restrict__ pack,
                                             int kt, int nt, int nt_total, int lane) {
  const bf16_t* p = pack + (((size_t)kt * nt_total + nt) * 32 + lane) * 16;
  union { v16bf v; uint4 u[2]; } r;
  r.u[0] = *reinterpret_cast<const uint4*>(p);
  r.u[1] = *reinterpret_cast<const uint4*>(p + 8);
  return r.v;
}

// fast silu: x * v_rcp_f32(1+exp(-x)) -- avoids the IEEE divide expansion
__device__ __forceinline__ float silu_f(float x) {
  return x * __builtin_amdgcn_rcpf(1.0f + __expf(-x));
}

// ---------------- weight packer: f32 [128 x wn] -> packed bf16 ----------------
__global__ void pack_w(const float* __restrict__ W, int wn,
                       bf16_t* __restrict__ dst, int nt_total, int nt_off) {
  int id = blockIdx.x * blockDim.x + threadIdx.x;
  int ntiles = wn >> 4;
  int total = 4 * ntiles * 32;
  if (id >= total) return;
  int lane = id & 31;
  int tile = (id >> 5) % ntiles;
  int kt   = (id >> 5) / ntiles;
  int n = tile * 16 + (lane & 15);
  int kbase = kt * 32 + ((lane & 16) ? 16 : 0);
  bf16_t* out = dst + (((size_t)kt * nt_total + (nt_off + tile)) * 32 + lane) * 16;
#pragma unroll
  for (int i = 0; i < 16; i++) out[i] = (bf16_t)W[(size_t)(kbase + i) * wn + n];
}

// ---------------- LayerNorm(x) -> bf16 ----------------
__global__ __launch_bounds__(128) void ln_kernel(const float* __restrict__ x,
                                                 const float* __restrict__ g,
                                                 const float* __restrict__ b,
                                                 bf16_t* __restrict__ xn_bf) {
  int n = blockIdx.x, h = threadIdx.x;
  __shared__ float sbuf[128];
  float v = x[(size_t)n * HH + h];
  sbuf[h] = v; __syncthreads();
  for (int s = 64; s > 0; s >>= 1) { if (h < s) sbuf[h] += sbuf[h + s]; __syncthreads(); }
  float mu = sbuf[0] * (1.0f / HH); __syncthreads();
  float dv = v - mu;
  sbuf[h] = dv * dv; __syncthreads();
  for (int s = 64; s > 0; s >>= 1) { if (h < s) sbuf[h] += sbuf[h + s]; __syncthreads(); }
  float var = sbuf[0] * (1.0f / HH);
  float o = dv * rsqrtf(var + 1e-5f) * g[h] + b[h];
  xn_bf[(size_t)n * HH + h] = (bf16_t)o;
}

// ---------------- elementwise converts ----------------
__global__ void cvt_bf(const float* __restrict__ in, bf16_t* __restrict__ out, size_t total) {
  size_t i = (size_t)blockIdx.x * blockDim.x + threadIdx.x;
  if (i < total) out[i] = (bf16_t)in[i];
}

__global__ void vec_prep(const float* __restrict__ vec, const float* __restrict__ vln_w,
                         float* __restrict__ vecw, bf16_t* __restrict__ vecw_bf, size_t total) {
  size_t i = (size_t)blockIdx.x * blockDim.x + threadIdx.x;
  if (i >= total) return;
  float v = vec[i] * vln_w[i & (HH - 1)];
  vecw[i] = v;
  vecw_bf[i] = (bf16_t)v;
}

// ---------------- generic GEMM: C[M,Ncols] = act(A[M,128] @ B + bias) ----------------
// Register-blocked: each wave computes one 16-row M-tile x 4 consecutive N-tiles
// (A fragment loaded once per k-step, 4 independent WMMA accumulation chains).
__global__ __launch_bounds__(256) void gemm_bf16_k128(
    const bf16_t* __restrict__ Abf, const bf16_t* __restrict__ Bp,
    const float* __restrict__ bias, float* __restrict__ C,
    int Mtiles, int Ntiles, int act) {
  int lane = threadIdx.x & 31;
  int wave = blockIdx.x * (blockDim.x >> 5) + (threadIdx.x >> 5);
  int ngroups = Ntiles >> 2;                 // Ntiles divisible by 4 for all calls
  if (wave >= Mtiles * ngroups) return;      // wave-uniform
  int mt = wave / ngroups;
  int nt0 = (wave % ngroups) * 4;
  const bf16_t* arow = Abf + (size_t)(mt * 16 + (lane & 15)) * HH;
  v8f acc[4] = {v8f{}, v8f{}, v8f{}, v8f{}};
#pragma unroll
  for (int kt = 0; kt < 4; ++kt) {
    if (kt < 3) __builtin_prefetch(arow + (kt + 1) * 32, 0, 3);
    v16bf a = load_a_row(arow, kt * 32, lane);
#pragma unroll
    for (int t = 0; t < 4; ++t)
      acc[t] = wmma_bf16(a, load_b_pack(Bp, kt, nt0 + t, Ntiles, lane), acc[t]);
  }
  int Ncols = Ntiles * 16;
  int rbase = mt * 16 + ((lane & 16) ? 8 : 0);
#pragma unroll
  for (int t = 0; t < 4; ++t) {
    int col = (nt0 + t) * 16 + (lane & 15);
    float bv = bias ? bias[col] : 0.0f;
#pragma unroll
    for (int r = 0; r < 8; r++) {
      float x = acc[t][r] + bv;
      if (act) x = silu_f(x);
      C[(size_t)(rbase + r) * Ncols + col] = x;
    }
  }
}

// ---------------- fused vec@W_vec: vec3 out + vec_dot = sum_l vec1*vec2 ----------------
// Tile rows = 2 nodes x 8 L-rows; each lane holds one (node,col) L-fiber in its 8 acc VGPRs.
__global__ __launch_bounds__(256) void vec_gemm_kernel(
    const bf16_t* __restrict__ Abf /*[N*L,128]*/, const bf16_t* __restrict__ Bp /*nt=24*/,
    float* __restrict__ vec3 /*[N*L,128]*/, float* __restrict__ vdot /*[N,128]*/) {
  int lane = threadIdx.x & 31;
  int wave = blockIdx.x * 8 + (threadIdx.x >> 5);
  int pair = wave >> 3;      // node pair
  int nt = wave & 7;         // column tile (within H)
  const bf16_t* arow = Abf + (size_t)(pair * 16 + (lane & 15)) * HH;
  v8f c1 = {}, c2 = {}, c3 = {};
#pragma unroll
  for (int kt = 0; kt < 4; ++kt) {
    if (kt < 3) __builtin_prefetch(arow + (kt + 1) * 32, 0, 3);
    v16bf a = load_a_row(arow, kt * 32, lane);
    c1 = wmma_bf16(a, load_b_pack(Bp, kt, nt,      24, lane), c1);
    c2 = wmma_bf16(a, load_b_pack(Bp, kt, nt + 8,  24, lane), c2);
    c3 = wmma_bf16(a, load_b_pack(Bp, kt, nt + 16, 24, lane), c3);
  }
  int col = nt * 16 + (lane & 15);
  int rbase = pair * 16 + ((lane & 16) ? 8 : 0);
  float dot = 0.0f;
#pragma unroll
  for (int r = 0; r < 8; r++) {
    dot += c1[r] * c2[r];
    vec3[(size_t)(rbase + r) * HH + col] = c3[r];
  }
  int node = pair * 2 + ((lane & 16) ? 1 : 0);
  vdot[(size_t)node * HH + col] = dot;
}

// ---------------- message: attn per head, vm ----------------
__global__ __launch_bounds__(256) void message_kernel(
    const float* __restrict__ qkv /*[N,384]*/, const float* __restrict__ dkdv /*[E,256]*/,
    const int* __restrict__ src, const int* __restrict__ dst,
    const float* __restrict__ r_ij, float* __restrict__ vm, bf16_t* __restrict__ vm_bf) {
  int tid = blockIdx.x * blockDim.x + threadIdx.x;
  int e = tid >> 7, h = tid & 127;
  int j = src[e], i = dst[e];
  float qv  = qkv[(size_t)i * 384 + h];
  float kv  = qkv[(size_t)j * 384 + 128 + h];
  float vv  = qkv[(size_t)j * 384 + 256 + h];
  float dk  = dkdv[(size_t)e * 256 + h];
  float dv  = dkdv[(size_t)e * 256 + 128 + h];
  float t = qv * kv * dk;
  // head dot-product: reduce over 16-lane segments (D=16) in-wave
  for (int m = 1; m < 16; m <<= 1) t += __shfl_xor(t, m, 32);
  float attn = silu_f(t);
  float r = r_ij[e];
  float cut = (r < 5.0f) ? 0.5f * (__cosf(0.6283185307179586f * r) + 1.0f) : 0.0f;
  attn *= cut;
  float vmv = vv * dv * attn;
  vm[(size_t)e * HH + h] = vmv;
  vm_bf[(size_t)e * HH + h] = (bf16_t)vmv;
}

// ---------------- scatter-add of vm and vec_m to target nodes ----------------
__global__ __launch_bounds__(256) void scatter_kernel(
    const float* __restrict__ vm, const float* __restrict__ s12 /*[E,256]*/,
    const float* __restrict__ vecw /*[N*L,128]*/, const float* __restrict__ d_ij /*[E,8]*/,
    const int* __restrict__ src, const int* __restrict__ dst,
    float* __restrict__ x_agg, float* __restrict__ vec_agg) {
  int tid = blockIdx.x * blockDim.x + threadIdx.x;
  int e = tid >> 7, h = tid & 127;
  int j = src[e], i = dst[e];
  float s1 = s12[(size_t)e * 256 + h];
  float s2 = s12[(size_t)e * 256 + 128 + h];
  atomicAdd(&x_agg[(size_t)i * HH + h], vm[(size_t)e * HH + h]);
#pragma unroll
  for (int l = 0; l < LL; l++) {
    float m = vecw[((size_t)j * LL + l) * HH + h] * s1 + s2 * d_ij[(size_t)e * LL + l];
    atomicAdd(&vec_agg[((size_t)i * LL + l) * HH + h], m);
  }
}

// ---------------- fused edge update: gathered WMMA GEMMs + rejection + w_dot + df ----------------
// Tile rows = 2 edges x 8 L-rows; rejection & dot are per-lane register math.
__global__ __launch_bounds__(256) void edge_w_kernel(
    const bf16_t* __restrict__ vecw_bf, const bf16_t* __restrict__ Btrg,
    const bf16_t* __restrict__ Bsrc, const int* __restrict__ src,
    const int* __restrict__ dst, const float* __restrict__ d_ij,
    const float* __restrict__ fsilu, float* __restrict__ df_out) {
  int lane = threadIdx.x & 31;
  int wave = blockIdx.x * 8 + (threadIdx.x >> 5);
  int pair = wave >> 3;   // edge pair
  int nt = wave & 7;      // column tile
  int row = lane & 15;
  int e_row = pair * 2 + (row >> 3);
  int l_row = row & 7;
  int n_d = dst[e_row], n_s = src[e_row];
  const bf16_t* a1 = vecw_bf + ((size_t)n_d * LL + l_row) * HH;  // vec[dst] rows
  const bf16_t* a2 = vecw_bf + ((size_t)n_s * LL + l_row) * HH;  // vec[src] rows
  v8f c1 = {}, c2 = {};
#pragma unroll
  for (int kt = 0; kt < 4; ++kt) {
    c1 = wmma_bf16(load_a_row(a1, kt * 32, lane), load_b_pack(Btrg, kt, nt, 8, lane), c1);
    c2 = wmma_bf16(load_a_row(a2, kt * 32, lane), load_b_pack(Bsrc, kt, nt, 8, lane), c2);
  }
  int my_e = pair * 2 + ((lane & 16) ? 1 : 0);
  const float* dp = d_ij + (size_t)my_e * LL;
  float d[8];
#pragma unroll
  for (int r = 0; r < 8; r++) d[r] = dp[r];
  // rejection: w' = w - (sum_l w*d) * d   (sign of d cancels, so src's -d_ij is equivalent)
  float p1 = 0.0f, p2 = 0.0f;
#pragma unroll
  for (int r = 0; r < 8; r++) { p1 += c1[r] * d[r]; p2 += c2[r] * d[r]; }
  float wdot = 0.0f;
#pragma unroll
  for (int r = 0; r < 8; r++) {
    float w1 = c1[r] - p1 * d[r];
    float w2 = c2[r] - p2 * d[r];
    wdot += w1 * w2;
  }
  int col = nt * 16 + (lane & 15);
  df_out[(size_t)my_e * HH + col] = fsilu[(size_t)my_e * HH + col] * wdot;
}

// ---------------- final node update ----------------
__global__ __launch_bounds__(256) void final_kernel(
    const float* __restrict__ o123 /*[N,384]*/, const float* __restrict__ vdot,
    const float* __restrict__ vec3 /*[N*L,128]*/, const float* __restrict__ vec_agg,
    float* __restrict__ dx, float* __restrict__ dvec) {
  int tid = blockIdx.x * blockDim.x + threadIdx.x;
  int n = tid >> 7, h = tid & 127;
  float o1 = o123[(size_t)n * 384 + h];
  float o2 = o123[(size_t)n * 384 + 128 + h];
  float o3 = o123[(size_t)n * 384 + 256 + h];
  dx[(size_t)n * HH + h] = vdot[(size_t)n * HH + h] * o2 + o3;
#pragma unroll
  for (int l = 0; l < LL; l++) {
    size_t idx = ((size_t)n * LL + l) * HH + h;
    dvec[idx] = vec3[idx] * o1 + vec_agg[idx];
  }
}

// ---------------- host orchestration ----------------
extern "C" void kernel_launch(void* const* d_in, const int* in_sizes, int n_in,
                              void* d_out, int out_size, void* d_ws, size_t ws_size,
                              hipStream_t stream) {
  constexpr int N = NN, E = EE, H = HH, L = LL;
  const float* x     = (const float*)d_in[0];
  const float* vec   = (const float*)d_in[1];
  const float* f_ij  = (const float*)d_in[2];
  const float* d_ij  = (const float*)d_in[3];
  const float* r_ij  = (const float*)d_in[4];
  const int*   eidx  = (const int*)  d_in[5];
  const float* ln_g  = (const float*)d_in[6];
  const float* ln_b  = (const float*)d_in[7];
  const float* vln_w = (const float*)d_in[8];
  const float* W_vec = (const float*)d_in[9];
  const float* W_q = (const float*)d_in[10]; const float* b_q = (const float*)d_in[11];
  const float* W_k = (const float*)d_in[12]; const float* b_k = (const float*)d_in[13];
  const float* W_v = (const float*)d_in[14]; const float* b_v = (const float*)d_in[15];
  const float* W_dk = (const float*)d_in[16]; const float* b_dk = (const float*)d_in[17];
  const float* W_dv = (const float*)d_in[18]; const float* b_dv = (const float*)d_in[19];
  const float* W_s = (const float*)d_in[20]; const float* b_s = (const float*)d_in[21];
  const float* W_f = (const float*)d_in[22]; const float* b_f = (const float*)d_in[23];
  const float* W_src = (const float*)d_in[24];
  const float* W_trg = (const float*)d_in[25];
  const float* W_o = (const float*)d_in[26]; const float* b_o = (const float*)d_in[27];

  const int* src = eidx;       // edge_index[0] = j (source)
  const int* dst = eidx + E;   // edge_index[1] = i (target)

  // outputs (flat, in return order): dx [N,H], dvec [N,L,H], df_ij [E,H]
  float* dx   = (float*)d_out;
  float* dvec = dx + (size_t)N * H;
  float* dfij = dvec + (size_t)N * L * H;

  // workspace carve-up (256B aligned)
  size_t off = 0;
  auto A = [&](size_t bytes) {
    size_t r = off; off = (off + bytes + 255) & ~(size_t)255;
    return (void*)((char*)d_ws + r);
  };
  bf16_t* xn_bf   = (bf16_t*)A((size_t)N * H * 2);
  bf16_t* f_bf    = (bf16_t*)A((size_t)E * H * 2);
  float*  vecw    = (float*) A((size_t)N * L * H * 4);
  bf16_t* vecw_bf = (bf16_t*)A((size_t)N * L * H * 2);
  float*  qkv     = (float*) A((size_t)N * 3 * H * 4);
  float*  dkdv    = (float*) A((size_t)E * 2 * H * 4);
  float*  fsilu   = (float*) A((size_t)E * H * 4);
  float*  vec3    = (float*) A((size_t)N * L * H * 4);
  float*  vdot    = (float*) A((size_t)N * H * 4);
  float*  vm      = (float*) A((size_t)E * H * 4);
  bf16_t* vm_bf   = (bf16_t*)A((size_t)E * H * 2);
  float*  s12     = (float*) A((size_t)E * 2 * H * 4);
  float*  xagg    = (float*) A((size_t)N * H * 4);
  bf16_t* xagg_bf = (bf16_t*)A((size_t)N * H * 2);
  float*  vagg    = (float*) A((size_t)N * L * H * 4);
  float*  o123    = (float*) A((size_t)N * 3 * H * 4);
  bf16_t* pQKV = (bf16_t*)A((size_t)4 * 24 * 32 * 16 * 2);
  bf16_t* pDKV = (bf16_t*)A((size_t)4 * 16 * 32 * 16 * 2);
  bf16_t* pVEC = (bf16_t*)A((size_t)4 * 24 * 32 * 16 * 2);
  bf16_t* pS   = (bf16_t*)A((size_t)4 * 16 * 32 * 16 * 2);
  bf16_t* pF   = (bf16_t*)A((size_t)4 *  8 * 32 * 16 * 2);
  bf16_t* pSRC = (bf16_t*)A((size_t)4 *  8 * 32 * 16 * 2);
  bf16_t* pTRG = (bf16_t*)A((size_t)4 *  8 * 32 * 16 * 2);
  bf16_t* pO   = (bf16_t*)A((size_t)4 * 24 * 32 * 16 * 2);
  float* bqkv = (float*)A(384 * 4);
  float* bdkv = (float*)A(256 * 4);

  auto pack = [&](const float* W, int wn, bf16_t* dstp, int ntt, int ntoff) {
    int total = 4 * (wn >> 4) * 32;
    pack_w<<<(total + 255) / 256, 256, 0, stream>>>(W, wn, dstp, ntt, ntoff);
  };
  // pack all weights (tiny, once per call)
  pack(W_q, 128, pQKV, 24, 0); pack(W_k, 128, pQKV, 24, 8); pack(W_v, 128, pQKV, 24, 16);
  pack(W_dk, 128, pDKV, 16, 0); pack(W_dv, 128, pDKV, 16, 8);
  pack(W_vec, 384, pVEC, 24, 0);
  pack(W_s, 256, pS, 16, 0);
  pack(W_f, 128, pF, 8, 0);
  pack(W_src, 128, pSRC, 8, 0);
  pack(W_trg, 128, pTRG, 8, 0);
  pack(W_o, 384, pO, 24, 0);
  // concat biases for fused GEMMs
  hipMemcpyAsync(bqkv,        b_q,  128 * 4, hipMemcpyDeviceToDevice, stream);
  hipMemcpyAsync(bqkv + 128,  b_k,  128 * 4, hipMemcpyDeviceToDevice, stream);
  hipMemcpyAsync(bqkv + 256,  b_v,  128 * 4, hipMemcpyDeviceToDevice, stream);
  hipMemcpyAsync(bdkv,        b_dk, 128 * 4, hipMemcpyDeviceToDevice, stream);
  hipMemcpyAsync(bdkv + 128,  b_dv, 128 * 4, hipMemcpyDeviceToDevice, stream);

  // prep: layernorm, bf16 converts, vec scale
  ln_kernel<<<N, 128, 0, stream>>>(x, ln_g, ln_b, xn_bf);
  cvt_bf<<<(int)(((size_t)E * H + 255) / 256), 256, 0, stream>>>(f_ij, f_bf, (size_t)E * H);
  vec_prep<<<(int)(((size_t)N * L * H + 255) / 256), 256, 0, stream>>>(vec, vln_w, vecw, vecw_bf, (size_t)N * L * H);

  auto gemm = [&](const bf16_t* Ap, const bf16_t* Bp, const float* bias, float* C,
                  int Mt, int Nt, int act) {
    int waves = Mt * (Nt >> 2);           // 4 N-tiles per wave
    gemm_bf16_k128<<<(waves + 7) / 8, 256, 0, stream>>>(Ap, Bp, bias, C, Mt, Nt, act);
  };
  gemm(xn_bf, pQKV, bqkv, qkv,   N / 16, 24, 0);   // q|k|v           [N,384]
  gemm(f_bf,  pDKV, bdkv, dkdv,  E / 16, 16, 1);   // silu(dk|dv)     [E,256]
  gemm(f_bf,  pF,   b_f,  fsilu, E / 16,  8, 1);   // silu(f@W_f+b_f) [E,128]

  // fused vec@W_vec -> vec3 + vec_dot
  vec_gemm_kernel<<<N / 2, 256, 0, stream>>>(vecw_bf, pVEC, vec3, vdot);

  // per-edge message + vm
  message_kernel<<<(E * H) / 256, 256, 0, stream>>>(qkv, dkdv, src, dst, r_ij, vm, vm_bf);
  gemm(vm_bf, pS, b_s, s12, E / 16, 16, 1);        // silu(vm@W_s+b_s) [E,256]

  // aggregate
  hipMemsetAsync(xagg, 0, (size_t)N * H * 4, stream);
  hipMemsetAsync(vagg, 0, (size_t)N * L * H * 4, stream);
  scatter_kernel<<<(E * H) / 256, 256, 0, stream>>>(vm, s12, vecw, d_ij, src, dst, xagg, vagg);

  // node output GEMM
  cvt_bf<<<(int)(((size_t)N * H + 255) / 256), 256, 0, stream>>>(xagg, xagg_bf, (size_t)N * H);
  gemm(xagg_bf, pO, b_o, o123, N / 16, 24, 0);     // o1|o2|o3 [N,384]

  // fused edge update -> df_ij (output)
  edge_w_kernel<<<(E / 2), 256, 0, stream>>>(vecw_bf, pTRG, pSRC, src, dst, d_ij, fsilu, dfij);

  // dx, dvec (outputs)
  final_kernel<<<(N * H) / 256, 256, 0, stream>>>(o123, vdot, vec3, vagg, dx, dvec);
}
